// MultiStepTaxiNet_36129264894613
// MI455X (gfx1250) — compile-verified, matrix operand order checked by reading, not run on hardware
//
#include <hip/hip_runtime.h>
#include <hip/hip_bf16.h>

typedef __attribute__((ext_vector_type(16))) _Float16 v16h;
typedef __attribute__((ext_vector_type(8)))  _Float16 v8h;
typedef __attribute__((ext_vector_type(8)))  float    v8f;

#define HSTRIDE 264   // halves per activation row (256 + 8 pad -> bank-conflict-free A reads)

// Packed-weight offsets in halves (each 32x16 tile = 512 halves = 1 KB)
#define O0 0          // layer0: 16 tiles (K padded 4->32)
#define O1 8192       // layer1: 128 tiles
#define O2 73728      // layer2: 128 tiles
#define O3 139264     // layer3: 128 tiles
#define O4 204800     // layer4: 8 tiles  (256 -> 16)
#define WP_TOTAL 208896  // total packed halves (417792 bytes in d_ws)

// ---------------------------------------------------------------------------
// Prep: convert f32 weights -> f16 packed B-fragment layout.
// Within a tile, lane l owns halves j=0..15 at byte offset l*32:
//   k = kt*32 + (l>=16 ? 8 : 0) + (j<8 ? j : j+8),  n = nt*16 + (l&15)
// (matches ISA 16-bit B 32x16 striping: lanes 0-15 K={0..7,16..23}, lanes 16-31 K={8..15,24..31})
// ---------------------------------------------------------------------------
__global__ __launch_bounds__(256) void pack_w(const float* __restrict__ W0,
                                              const float* __restrict__ W1,
                                              const float* __restrict__ W2,
                                              const float* __restrict__ W3,
                                              const float* __restrict__ W4,
                                              _Float16* __restrict__ wp) {
    int tid = blockIdx.x * blockDim.x + threadIdx.x;
    if (tid >= WP_TOTAL) return;
    const float* W; int off, KT, Kact, N;
    if      (tid < O1) { W = W0; off = O0; KT = 1; Kact = 4;   N = 256; }
    else if (tid < O2) { W = W1; off = O1; KT = 8; Kact = 256; N = 256; }
    else if (tid < O3) { W = W2; off = O2; KT = 8; Kact = 256; N = 256; }
    else if (tid < O4) { W = W3; off = O3; KT = 8; Kact = 256; N = 256; }
    else               { W = W4; off = O4; KT = 8; Kact = 256; N = 16;  }
    int li     = tid - off;
    int tile   = li >> 9;
    int within = li & 511;
    int lane   = within >> 4;
    int j      = within & 15;
    int nt     = tile / KT;
    int kt     = tile - nt * KT;
    int k = kt * 32 + ((lane >= 16) ? 8 : 0) + ((j < 8) ? j : (j + 8));
    int n = nt * 16 + (lane & 15);
    float v = (k < Kact) ? W[k * N + n] : 0.0f;
    wp[tid] = (_Float16)v;
}

// ---------------------------------------------------------------------------
// Main kernel helpers
// ---------------------------------------------------------------------------
__device__ __forceinline__ float reluf(float v) { return v > 0.0f ? v : 0.0f; }

__device__ __forceinline__ v8f wmma_f16(v16h a, v16h b, v8f c) {
    return __builtin_amdgcn_wmma_f32_16x16x32_f16(false, a, false, b, (short)0, c, false, false);
}

// C fragment (8 f32 VGPRs; lane=col, VGPR j=row j + 8*(lane>=16)) -> bias+ReLU -> f16 row-major LDS
__device__ __forceinline__ void store_tile(_Float16* hw, v8f c, float bias, int mbase, int col, int lane) {
    int rbase = mbase + ((lane >> 4) << 3);
#pragma unroll
    for (int j = 0; j < 8; ++j)
        hw[(rbase + j) * HSTRIDE + col] = (_Float16)reluf(c[j] + bias);
}

// Load all A fragments (2 M-tiles x 8 K-tiles) from row-major f16 LDS into VGPRs.
// A 16x32 f16 layout: lane = row (l&15); lanes<16 hold K {0..7,16..23}, lanes>=16 {8..15,24..31}
__device__ __forceinline__ void load_A(const _Float16* hw, int lane, v16h A[2][8]) {
    int col  = lane & 15;
    int ksub = (lane >> 4) << 3;
#pragma unroll
    for (int t = 0; t < 2; ++t) {
#pragma unroll
        for (int kt = 0; kt < 8; ++kt) {
            const _Float16* base = hw + (t * 16 + col) * HSTRIDE + kt * 32 + ksub;
            v8h lo = *(const v8h*)(base);
            v8h hi = *(const v8h*)(base + 16);
            v16h a;
#pragma unroll
            for (int j = 0; j < 8; ++j) { a[j] = lo[j]; a[j + 8] = hi[j]; }
            A[t][kt] = a;
        }
    }
}

__device__ __forceinline__ void layer_256_256(_Float16* hw, const _Float16* __restrict__ wl,
                                              const float* __restrict__ bvec, int lane) {
    v16h A[2][8];
    load_A(hw, lane, A);
    int col = lane & 15;
    for (int nt = 0; nt < 16; ++nt) {
        float bias = bvec[nt * 16 + col];
        v8f c0 = {}, c1 = {};
#pragma unroll
        for (int kt = 0; kt < 8; ++kt) {
            v16h b = *(const v16h*)(wl + (nt * 8 + kt) * 512 + lane * 16);
            c0 = wmma_f16(A[0][kt], b, c0);
            c1 = wmma_f16(A[1][kt], b, c1);
        }
        store_tile(hw, c0, bias, 0,  nt * 16 + col, lane);
        store_tile(hw, c1, bias, 16, nt * 16 + col, lane);
    }
}

// ---------------------------------------------------------------------------
// Main kernel: 8 waves/WG, each wave owns 32 batch rows (two 16-row M-tiles)
// ---------------------------------------------------------------------------
__global__ __launch_bounds__(256) void taxinet_kernel(
    const float* __restrict__ x, const _Float16* __restrict__ wp,
    const float* __restrict__ b0v, const float* __restrict__ b1v,
    const float* __restrict__ b2v, const float* __restrict__ b3v,
    const float* __restrict__ b4v,
    const float* __restrict__ W5, const float* __restrict__ b5,
    const float* __restrict__ W6, const float* __restrict__ b6,
    const float* __restrict__ W7, const float* __restrict__ b7,
    const float* __restrict__ W8,
    float* __restrict__ out)
{
    extern __shared__ char smem[];
    _Float16* hbuf   = (_Float16*)smem;              // 8 * 32*264 halves = 135168 B
    float*    h4buf  = (float*)(smem + 135168);      // 8 * 32*17 f32     =  17408 B
    float*    inpbuf = (float*)(smem + 152576);      // 8 * 32*4  f32     =   4096 B
    float*    sw     = (float*)(smem + 156672);      // 256 f32 (small-layer weights)

    const int tid  = threadIdx.x;
    const int lane = tid & 31;
    const int wave = tid >> 5;
    const int col  = lane & 15;

    // Stage tiny layer-5..8 weights into LDS (broadcast reads later)
    if (tid < 228) {
        float v;
        if      (tid < 128) v = W5[tid];
        else if (tid < 136) v = b5[tid - 128];
        else if (tid < 200) v = W6[tid - 136];
        else if (tid < 208) v = b6[tid - 200];
        else if (tid < 224) v = W7[tid - 208];
        else if (tid < 226) v = b7[tid - 224];
        else                v = W8[tid - 226];
        sw[tid] = v;
    }
    __syncthreads();

    _Float16* hw  = hbuf   + wave * (32 * HSTRIDE);
    float*    h4w = h4buf  + wave * (32 * 17);
    float*    inw = inpbuf + wave * (32 * 4);

    const int    row = blockIdx.x * 256 + wave * 32 + lane;
    const float* xr  = x + (size_t)row * 10;
    float p  = xr[0];
    float th = xr[1];

    for (int s = 0; s < 4; ++s) {
        // -------- stage per-row 4-wide MLP input --------
        inw[lane * 4 + 0] = xr[2 + 2 * s];
        inw[lane * 4 + 1] = xr[3 + 2 * s];
        inw[lane * 4 + 2] = p  * (1.0f / 6.366468343804353f);
        inw[lane * 4 + 3] = th * (1.0f / 17.248858791583547f);
        __syncthreads();

        // -------- layer 0 (4 -> 256, K zero-padded to 32) --------
        v16h a0[2];
#pragma unroll
        for (int t = 0; t < 2; ++t) {
            v16h a = {};
            if (lane < 16) {    // lanes<16 own K=0..7 -> first 4 are real inputs
                const float* ip = inw + (t * 16 + lane) * 4;
                a[0] = (_Float16)ip[0]; a[1] = (_Float16)ip[1];
                a[2] = (_Float16)ip[2]; a[3] = (_Float16)ip[3];
            }
            a0[t] = a;
        }
        for (int nt = 0; nt < 16; ++nt) {
            float bias = b0v[nt * 16 + col];
            v16h b = *(const v16h*)(wp + O0 + nt * 512 + lane * 16);
            v8f c0 = {}, c1 = {};
            c0 = wmma_f16(a0[0], b, c0);
            c1 = wmma_f16(a0[1], b, c1);
            store_tile(hw, c0, bias, 0,  nt * 16 + col, lane);
            store_tile(hw, c1, bias, 16, nt * 16 + col, lane);
        }

        // -------- layers 1-3 (256 -> 256) --------
        layer_256_256(hw, wp + O1, b1v, lane);
        layer_256_256(hw, wp + O2, b2v, lane);
        layer_256_256(hw, wp + O3, b3v, lane);

        // -------- layer 4 (256 -> 16) --------
        {
            v16h A[2][8];
            load_A(hw, lane, A);
            float bias = b4v[col];
            v8f c0 = {}, c1 = {};
#pragma unroll
            for (int kt = 0; kt < 8; ++kt) {
                v16h b = *(const v16h*)(wp + O4 + kt * 512 + lane * 16);
                c0 = wmma_f16(A[0][kt], b, c0);
                c1 = wmma_f16(A[1][kt], b, c1);
            }
            int rbase = (lane >> 4) << 3;
#pragma unroll
            for (int j = 0; j < 8; ++j) {
                h4w[(rbase + j) * 17 + col]        = reluf(c0[j] + bias);
                h4w[(16 + rbase + j) * 17 + col]   = reluf(c1[j] + bias);
            }
        }

        // -------- tail: layers 5-8 + physics, one row per lane --------
        float hv[16];
#pragma unroll
        for (int i = 0; i < 16; ++i) hv[i] = h4w[lane * 17 + i];

        float a5[8];
#pragma unroll
        for (int n = 0; n < 8; ++n) {
            float acc = sw[128 + n];
#pragma unroll
            for (int k = 0; k < 16; ++k) acc += hv[k] * sw[k * 8 + n];
            a5[n] = reluf(acc);
        }
        float a6[8];
#pragma unroll
        for (int n = 0; n < 8; ++n) {
            float acc = sw[200 + n];
#pragma unroll
            for (int k = 0; k < 8; ++k) acc += a5[k] * sw[136 + k * 8 + n];
            a6[n] = reluf(acc);
        }
        float o0 = sw[224], o1 = sw[225];
#pragma unroll
        for (int k = 0; k < 8; ++k) {
            o0 += a6[k] * sw[208 + k * 2];
            o1 += a6[k] * sw[208 + k * 2 + 1];
        }
        float u   = o0 * sw[226] + o1 * sw[227];
        float dth = 2.8647889756541161f * __tanf(u * 0.017453292519943295f); // 0.05*180/pi * tan(u*pi/180)
#pragma unroll
        for (int it = 0; it < 20; ++it) {
            p  += 0.25f * __sinf(th * 0.017453292519943295f);
            th += dth;
        }
        __syncthreads();   // step rollover: inp / h4 regions reused next step
    }

    out[row] = p;   // INDEX == 0 -> p only
}

// ---------------------------------------------------------------------------
extern "C" void kernel_launch(void* const* d_in, const int* in_sizes, int n_in,
                              void* d_out, int out_size, void* d_ws, size_t ws_size,
                              hipStream_t stream) {
    const float* x  = (const float*)d_in[0];
    const float* W0 = (const float*)d_in[1];
    const float* b0 = (const float*)d_in[2];
    const float* W1 = (const float*)d_in[3];
    const float* b1 = (const float*)d_in[4];
    const float* W2 = (const float*)d_in[5];
    const float* b2 = (const float*)d_in[6];
    const float* W3 = (const float*)d_in[7];
    const float* b3 = (const float*)d_in[8];
    const float* W4 = (const float*)d_in[9];
    const float* b4 = (const float*)d_in[10];
    const float* W5 = (const float*)d_in[11];
    const float* b5 = (const float*)d_in[12];
    const float* W6 = (const float*)d_in[13];
    const float* b6 = (const float*)d_in[14];
    const float* W7 = (const float*)d_in[15];
    const float* b7 = (const float*)d_in[16];
    const float* W8 = (const float*)d_in[17];

    _Float16* wp = (_Float16*)d_ws;   // 417792 bytes of packed f16 weights

    pack_w<<<(WP_TOTAL + 255) / 256, 256, 0, stream>>>(W0, W1, W2, W3, W4, wp);

    const int B      = in_sizes[0] / 10;   // 131072
    const int blocks = B / 256;            // 256 rows per workgroup
    const size_t smem_bytes = 157696;
    taxinet_kernel<<<blocks, 256, smem_bytes, stream>>>(
        x, wp, b0, b1, b2, b3, b4, W5, b5, W6, b6, W7, b7, W8, (float*)d_out);
}
